// GPT_35210141892780
// MI455X (gfx1250) — matrix operand order repaired and checked by reading, not compiled, not run on hardware
//
#include <hip/hip_runtime.h>

// ---------------------------------------------------------------------------
// GPT decoder forward for gfx1250 (MI455X). f16 WMMA GEMMs w/ fp32 accum.
// Wave tile 32x32 (4 wmma accum chains), block tile 128x64 (8 waves).
// ---------------------------------------------------------------------------

typedef __attribute__((ext_vector_type(16))) _Float16 v16h;
typedef __attribute__((ext_vector_type(8)))  _Float16 v8h;
typedef __attribute__((ext_vector_type(8)))  float    v8f;

constexpr int kB  = 2;
constexpr int kT  = 512;
constexpr int kTA = 512;
constexpr int kC  = 1024;
constexpr int kH  = 16;
constexpr int kHD = 64;
constexpr int kL  = 6;
constexpr int kF  = 4096;
constexpr int kV  = 32000;
constexpr int kBT  = kB * kT;   // 1024
constexpr int kBTA = kB * kTA;  // 1024

__device__ __forceinline__ float gelu_tanh(float x) {
  float x3 = x * x * x;
  return 0.5f * x * (1.0f + tanhf(0.7978845608028654f * (x + 0.044715f * x3)));
}

// ---------------------------------------------------------------------------
// Embedding: x[b*T+t, :] = wte[idx[b,t], :] + wpe[t, :]
// ---------------------------------------------------------------------------
__global__ __launch_bounds__(256)
void embed_kernel(const int* __restrict__ idx, const float* __restrict__ wte,
                  const float* __restrict__ wpe, float* __restrict__ X) {
  int bt = blockIdx.x;
  int t  = bt % kT;
  int tok = idx[bt];
  const float* we = wte + (size_t)tok * kC;
  const float* wp = wpe + (size_t)t * kC;
  float* o = X + (size_t)bt * kC;
  for (int c = threadIdx.x; c < kC; c += 256) o[c] = we[c] + wp[c];
}

// ---------------------------------------------------------------------------
// fp32 -> f16 elementwise (for xa)
// ---------------------------------------------------------------------------
__global__ __launch_bounds__(256)
void cvt_f16_kernel(const float* __restrict__ X, _Float16* __restrict__ O, int n) {
  int i = blockIdx.x * 256 + threadIdx.x;
  if (i < n) O[i] = (_Float16)X[i];
}

// ---------------------------------------------------------------------------
// Weight transpose + convert: W[K,N] fp32 -> Wt[N,K] f16 (so GEMM B-operand
// streams contiguously along K and loads with global_load_b128)
// ---------------------------------------------------------------------------
__global__ __launch_bounds__(256)
void wt16_kernel(const float* __restrict__ W, _Float16* __restrict__ Wt, int K, int N) {
  int o = blockIdx.x * 256 + threadIdx.x;
  if (o >= K * N) return;
  int n = o / K;
  int k = o - n * K;
  Wt[o] = (_Float16)W[(size_t)k * N + n];
}

// ---------------------------------------------------------------------------
// LayerNorm (one row per workgroup) -> f16 output for WMMA GEMM input
// ---------------------------------------------------------------------------
__global__ __launch_bounds__(256)
void ln_f16_kernel(const float* __restrict__ X, const float* __restrict__ w,
                   const float* __restrict__ b, _Float16* __restrict__ O) {
  int row = blockIdx.x;
  const float* x = X + (size_t)row * kC;
  float s = 0.f, s2 = 0.f;
  for (int c = threadIdx.x; c < kC; c += 256) { float v = x[c]; s += v; s2 += v * v; }
  __shared__ float rs[256], rq[256];
  rs[threadIdx.x] = s; rq[threadIdx.x] = s2;
  __syncthreads();
  for (int off = 128; off; off >>= 1) {
    if (threadIdx.x < off) { rs[threadIdx.x] += rs[threadIdx.x + off]; rq[threadIdx.x] += rq[threadIdx.x + off]; }
    __syncthreads();
  }
  float mean = rs[0] * (1.0f / kC);
  float var  = rq[0] * (1.0f / kC) - mean * mean;
  float rinv = rsqrtf(var + 1e-5f);
  _Float16* o = O + (size_t)row * kC;
  for (int c = threadIdx.x; c < kC; c += 256)
    o[c] = (_Float16)((x[c] - mean) * rinv * w[c] + b[c]);
}

// ---------------------------------------------------------------------------
// Causal softmax over scores [Z, T, Tk] fp32 -> probs f16 (scale fused)
// ---------------------------------------------------------------------------
__global__ __launch_bounds__(256)
void softmax_kernel(const float* __restrict__ S, _Float16* __restrict__ P,
                    int Tk, float scale) {
  int row = blockIdx.x;
  int z   = blockIdx.y;
  size_t base = ((size_t)z * gridDim.x + row) * Tk;
  int nvalid = row + 1;  // tril mask (applies to cross-attn too, per reference)
  float mx = -1e30f;
  for (int c = threadIdx.x; c < nvalid; c += 256) mx = fmaxf(mx, S[base + c] * scale);
  __shared__ float red[256];
  red[threadIdx.x] = mx;
  __syncthreads();
  for (int off = 128; off; off >>= 1) {
    if (threadIdx.x < off) red[threadIdx.x] = fmaxf(red[threadIdx.x], red[threadIdx.x + off]);
    __syncthreads();
  }
  mx = red[0];
  __syncthreads();
  float sum = 0.f;
  for (int c = threadIdx.x; c < nvalid; c += 256) sum += __expf(S[base + c] * scale - mx);
  red[threadIdx.x] = sum;
  __syncthreads();
  for (int off = 128; off; off >>= 1) {
    if (threadIdx.x < off) red[threadIdx.x] += red[threadIdx.x + off];
    __syncthreads();
  }
  float inv = 1.0f / red[0];
  for (int c = threadIdx.x; c < Tk; c += 256) {
    float p = (c < nvalid) ? __expf(S[base + c] * scale - mx) * inv : 0.0f;
    P[base + c] = (_Float16)p;
  }
}

// ---------------------------------------------------------------------------
// Pack V fp32 [B, Tsrc, C] head-slices into Vt f16 [B, H, HD, Tsrc]
// ---------------------------------------------------------------------------
__global__ __launch_bounds__(256)
void pack_vt_kernel(const float* __restrict__ V32, _Float16* __restrict__ VT, int Tsrc) {
  int i = blockIdx.x * 256 + threadIdx.x;  // i = ((b*H + h)*HD + d)*Tsrc + t
  int t  = i % Tsrc;
  int d  = (i / Tsrc) % kHD;
  int hh = (i / (Tsrc * kHD)) % kH;
  int bb = i / (Tsrc * kHD * kH);
  VT[i] = (_Float16)V32[((size_t)(bb * Tsrc + t)) * kC + hh * kHD + d];
}

// ---------------------------------------------------------------------------
// Batched WMMA GEMM: D[z] = A[z] (f16, [M,K]) * Bt[z]^T (f16/f32, [N,K]) (+bias)
// (+residual) (+gelu), outputs fp32 D and/or f16 Dh.
// Block = 256 threads = 8 waves arranged 4(M) x 2(N): block tile 128x64.
// Wave tile 32x32: 2 A-fragments x 2 B-fragments -> 4 independent
// v_wmma_f32_16x16x32_f16 accumulate chains (2 b128 loads per WMMA).
// z decomposed as (bI, hI) = (z / Hn, z % Hn) with separate strides so
// per-head slices of [B,T,C] tensors batch without repacking.
// ---------------------------------------------------------------------------
__global__ __launch_bounds__(256)
void gemm_f16_wmma(const _Float16* __restrict__ A, long long a_sb, long long a_sh, int lda,
                   const void* __restrict__ Bv, long long b_sb, long long b_sh, int ldb, int b_f32,
                   const float* __restrict__ bias,
                   const float* __restrict__ R, long long r_sb, long long r_sh, int ldr,
                   float* __restrict__ D, long long d_sb, long long d_sh, int ldd,
                   _Float16* __restrict__ Dh, long long dh_sb, long long dh_sh, int lddh,
                   int M, int N, int K, int Hn, int act_gelu) {
  const int lane = threadIdx.x & 31;
  const int wave = threadIdx.x >> 5;
  const int wm = wave >> 1, wn = wave & 1;
  const int z  = blockIdx.z;
  const int bI = z / Hn, hI = z - bI * Hn;

  const int tm = blockIdx.x * 128 + wm * 32;
  const int tn = blockIdx.y * 64 + wn * 32;

  const int kbA  = (lane & 16) ? 8 : 0;    // A 16x32 f16 layout K-split (ISA 7.12.2)
  const int col0 = tn + (lane & 15);
  const int col1 = col0 + 16;
  const int kbB  = (lane & 16) ? 8 : 0;    // B 32x16 f16 layout K-split

  const _Float16* Ab0 =
      A + (size_t)bI * a_sb + (size_t)hI * a_sh + (size_t)(tm + (lane & 15)) * lda + kbA;
  const _Float16* Ab1 = Ab0 + (size_t)16 * lda;

  v8f c00 = {}, c01 = {}, c10 = {}, c11 = {};

  if (!b_f32) {
    const _Float16* B16 = (const _Float16*)Bv + (size_t)bI * b_sb + (size_t)hI * b_sh + kbB;
    const _Float16* Bp0 = B16 + (size_t)col0 * ldb;
    const _Float16* Bp1 = B16 + (size_t)col1 * ldb;
    for (int kk = 0; kk < K; kk += 32) {
      if (kk + 32 < K) {  // pull next K-tile of the weight stream into cache
        __builtin_prefetch(Bp0 + kk + 32, 0, 1);
        __builtin_prefetch(Bp1 + kk + 32, 0, 1);
      }
      v8h a00 = *(const v8h*)(Ab0 + kk);
      v8h a01 = *(const v8h*)(Ab0 + kk + 16);
      v8h a10 = *(const v8h*)(Ab1 + kk);
      v8h a11 = *(const v8h*)(Ab1 + kk + 16);
      v8h x0  = *(const v8h*)(Bp0 + kk);
      v8h x1  = *(const v8h*)(Bp0 + kk + 16);
      v8h x2  = *(const v8h*)(Bp1 + kk);
      v8h x3  = *(const v8h*)(Bp1 + kk + 16);
      v16h a0, a1, b0, b1;
      #pragma unroll
      for (int i = 0; i < 8; ++i) {
        a0[i] = a00[i]; a0[8 + i] = a01[i];
        a1[i] = a10[i]; a1[8 + i] = a11[i];
        b0[i] = x0[i];  b0[8 + i] = x1[i];
        b1[i] = x2[i];  b1[8 + i] = x3[i];
      }
      c00 = __builtin_amdgcn_wmma_f32_16x16x32_f16(false, a0, false, b0, (short)0, c00, false, false);
      c01 = __builtin_amdgcn_wmma_f32_16x16x32_f16(false, a0, false, b1, (short)0, c01, false, false);
      c10 = __builtin_amdgcn_wmma_f32_16x16x32_f16(false, a1, false, b0, (short)0, c10, false, false);
      c11 = __builtin_amdgcn_wmma_f32_16x16x32_f16(false, a1, false, b1, (short)0, c11, false, false);
    }
  } else {
    // fp32 B path (lm_head: wte [V,C] is already [N,K] layout) — convert in regs
    const float* B32 = (const float*)Bv + (size_t)bI * b_sb + (size_t)hI * b_sh + kbB;
    const float* Bp0 = B32 + (size_t)col0 * ldb;
    const float* Bp1 = B32 + (size_t)col1 * ldb;
    for (int kk = 0; kk < K; kk += 32) {
      if (kk + 32 < K) {
        __builtin_prefetch(Bp0 + kk + 32, 0, 1);
        __builtin_prefetch(Bp1 + kk + 32, 0, 1);
      }
      v8h a00 = *(const v8h*)(Ab0 + kk);
      v8h a01 = *(const v8h*)(Ab0 + kk + 16);
      v8h a10 = *(const v8h*)(Ab1 + kk);
      v8h a11 = *(const v8h*)(Ab1 + kk + 16);
      v8f f0  = *(const v8f*)(Bp0 + kk);
      v8f f1  = *(const v8f*)(Bp0 + kk + 16);
      v8f f2  = *(const v8f*)(Bp1 + kk);
      v8f f3  = *(const v8f*)(Bp1 + kk + 16);
      v16h a0, a1, b0, b1;
      #pragma unroll
      for (int i = 0; i < 8; ++i) {
        a0[i] = a00[i]; a0[8 + i] = a01[i];
        a1[i] = a10[i]; a1[8 + i] = a11[i];
        b0[i] = (_Float16)f0[i]; b0[8 + i] = (_Float16)f1[i];
        b1[i] = (_Float16)f2[i]; b1[8 + i] = (_Float16)f3[i];
      }
      c00 = __builtin_amdgcn_wmma_f32_16x16x32_f16(false, a0, false, b0, (short)0, c00, false, false);
      c01 = __builtin_amdgcn_wmma_f32_16x16x32_f16(false, a0, false, b1, (short)0, c01, false, false);
      c10 = __builtin_amdgcn_wmma_f32_16x16x32_f16(false, a1, false, b0, (short)0, c10, false, false);
      c11 = __builtin_amdgcn_wmma_f32_16x16x32_f16(false, a1, false, b1, (short)0, c11, false, false);
    }
  }

  // Epilogue: C/D fragment layout — VGPR v holds row base+v (lanes 0-15) /
  // base+v+8 (lanes 16-31), column = tile base + (lane&15).
  const size_t roff  = R  ? ((size_t)bI * r_sb  + (size_t)hI * r_sh)  : 0;
  const size_t doff  = D  ? ((size_t)bI * d_sb  + (size_t)hI * d_sh)  : 0;
  const size_t dhoff = Dh ? ((size_t)bI * dh_sb + (size_t)hI * dh_sh) : 0;
  const int rhalf = (lane & 16) ? 8 : 0;
  #pragma unroll
  for (int mi = 0; mi < 2; ++mi) {
    const int rbase = tm + mi * 16 + rhalf;
    v8f q0 = mi ? c10 : c00;
    v8f q1 = mi ? c11 : c01;
    #pragma unroll
    for (int v = 0; v < 8; ++v) {
      int row = rbase + v;
      float y0 = q0[v], y1 = q1[v];
      if (bias) { y0 += bias[col0]; y1 += bias[col1]; }
      if (R) {
        y0 += R[roff + (size_t)row * ldr + col0];
        y1 += R[roff + (size_t)row * ldr + col1];
      }
      if (act_gelu) { y0 = gelu_tanh(y0); y1 = gelu_tanh(y1); }
      if (D) {
        D[doff + (size_t)row * ldd + col0] = y0;
        D[doff + (size_t)row * ldd + col1] = y1;
      }
      if (Dh) {
        Dh[dhoff + (size_t)row * lddh + col0] = (_Float16)y0;
        Dh[dhoff + (size_t)row * lddh + col1] = (_Float16)y1;
      }
    }
  }
}

// ---------------------------------------------------------------------------
// Host orchestration
// ---------------------------------------------------------------------------
extern "C" void kernel_launch(void* const* d_in, const int* in_sizes, int n_in,
                              void* d_out, int out_size, void* d_ws, size_t ws_size,
                              hipStream_t stream) {
  const int*   idx  = (const int*)  d_in[0];
  const float* xa   = (const float*)d_in[1];
  const float* wte  = (const float*)d_in[2];
  const float* wpe  = (const float*)d_in[3];
  const float* ln1w = (const float*)d_in[4];
  const float* ln1b = (const float*)d_in[5];
  const float* sawq = (const float*)d_in[6];
  const float* sabq = (const float*)d_in[7];
  const float* sawk = (const float*)d_in[8];
  const float* sabk = (const float*)d_in[9];
  const float* sawv = (const float*)d_in[10];
  const float* sabv = (const float*)d_in[11];
  const float* sawo = (const float*)d_in[12];
  const float* sabo = (const float*)d_in[13];
  const float* cawq = (const float*)d_in[14];
  const float* cabq = (const float*)d_in[15];
  const float* cawk = (const float*)d_in[16];
  const float* cabk = (const float*)d_in[17];
  const float* cawv = (const float*)d_in[18];
  const float* cabv = (const float*)d_in[19];
  const float* cawo = (const float*)d_in[20];
  const float* cabo = (const float*)d_in[21];
  const float* ln2w = (const float*)d_in[22];
  const float* ln2b = (const float*)d_in[23];
  const float* fcw  = (const float*)d_in[24];
  const float* fcb  = (const float*)d_in[25];
  const float* prw  = (const float*)d_in[26];
  const float* prb  = (const float*)d_in[27];
  const float* lnfw = (const float*)d_in[28];
  const float* lnfb = (const float*)d_in[29];

  char* wp = (char*)d_ws;
  auto take = [&](size_t bytes) -> void* {
    void* r = wp;
    wp += (bytes + 255) & ~(size_t)255;
    return r;
  };
  float*    h    = (float*)   take((size_t)kBT  * kC * 4);  // residual stream fp32
  _Float16* n16  = (_Float16*)take((size_t)kBT  * kC * 2);  // LN output f16
  _Float16* q16  = (_Float16*)take((size_t)kBT  * kC * 2);
  _Float16* k16  = (_Float16*)take((size_t)kBTA * kC * 2);
  float*    v32  = (float*)   take((size_t)kBTA * kC * 4);
  _Float16* vt16 = (_Float16*)take((size_t)kB * kH * kHD * kT * 2);
  _Float16* y16  = (_Float16*)take((size_t)kBT  * kC * 2);
  float*    sc   = (float*)   take((size_t)kB * kH * kT * kT * 4);
  _Float16* pr16 = (_Float16*)take((size_t)kB * kH * kT * kT * 2);
  _Float16* m16  = (_Float16*)take((size_t)kBT * kF * 2);
  _Float16* xa16 = (_Float16*)take((size_t)kBTA * kC * 2);
  _Float16* wbuf = (_Float16*)take((size_t)kF * kC * 2);   // staged transposed weight

  auto gemm = [&](const _Float16* A, long long asb, long long ash, int lda,
                  const void* Bp, long long bsb, long long bsh, int ldb, int bf32,
                  const float* bias,
                  const float* R, long long rsb, long long rsh, int ldr,
                  float* Dst, long long dsb, long long dsh, int ldd,
                  _Float16* DstH, long long dhsb, long long dhsh, int lddh,
                  int M, int N, int K, int batch, int Hn, int gelu) {
    dim3 grid(M / 128, N / 64, batch);
    gemm_f16_wmma<<<grid, 256, 0, stream>>>(A, asb, ash, lda, Bp, bsb, bsh, ldb, bf32,
                                            bias, R, rsb, rsh, ldr,
                                            Dst, dsb, dsh, ldd, DstH, dhsb, dhsh, lddh,
                                            M, N, K, Hn, gelu);
  };
  auto wt = [&](const float* W, int K, int N) {  // stage Wt[N,K] f16 into wbuf
    int tot = K * N;
    wt16_kernel<<<(tot + 255) / 256, 256, 0, stream>>>(W, wbuf, K, N);
  };

  // x = wte[idx] + wpe[:T]
  embed_kernel<<<kBT, 256, 0, stream>>>(idx, wte, wpe, h);
  cvt_f16_kernel<<<(kBTA * kC) / 256, 256, 0, stream>>>(xa, xa16, kBTA * kC);

  auto attention = [&](const _Float16* nq, const _Float16* nkv, int Tkv,
                       const float* wq_, const float* bq_, const float* wk_, const float* bk_,
                       const float* wv_, const float* bv_, const float* wo_, const float* bo_) {
    int Mkv = kB * Tkv;
    // q, k (f16 outputs), v (fp32 output -> transposed pack)
    wt(wq_, kC, kC);
    gemm(nq, 0, 0, kC, wbuf, 0, 0, kC, 0, bq_, nullptr, 0, 0, 0,
         nullptr, 0, 0, 0, q16, 0, 0, kC, kBT, kC, kC, 1, 1, 0);
    wt(wk_, kC, kC);
    gemm(nkv, 0, 0, kC, wbuf, 0, 0, kC, 0, bk_, nullptr, 0, 0, 0,
         nullptr, 0, 0, 0, k16, 0, 0, kC, Mkv, kC, kC, 1, 1, 0);
    wt(wv_, kC, kC);
    gemm(nkv, 0, 0, kC, wbuf, 0, 0, kC, 0, bv_, nullptr, 0, 0, 0,
         v32, 0, 0, kC, nullptr, 0, 0, 0, Mkv, kC, kC, 1, 1, 0);
    pack_vt_kernel<<<(kB * kH * kHD * Tkv) / 256, 256, 0, stream>>>(v32, vt16, Tkv);
    // scores[b,h] = Q[b,:,h,:] @ K[b,:,h,:]^T   (K already in [N,K] layout)
    gemm(q16, (long long)kT * kC, kHD, kC,
         k16, (long long)Tkv * kC, kHD, kC, 0, nullptr,
         nullptr, 0, 0, 0,
         sc, (long long)kH * kT * Tkv, (long long)kT * Tkv, Tkv,
         nullptr, 0, 0, 0,
         kT, Tkv, kHD, kB * kH, kH, 0);
    dim3 sg(kT, kB * kH);
    softmax_kernel<<<sg, 256, 0, stream>>>(sc, pr16, Tkv, 0.125f);
    // y[b,:,h,:] = P[b,h] @ V[b,:,h,:]
    gemm(pr16, (long long)kH * kT * Tkv, (long long)kT * Tkv, Tkv,
         vt16, (long long)kH * kHD * Tkv, (long long)kHD * Tkv, Tkv, 0, nullptr,
         nullptr, 0, 0, 0,
         nullptr, 0, 0, 0,
         y16, (long long)kT * kC, kHD, kC,
         kT, kHD, Tkv, kB * kH, kH, 0);
    // h += y @ wo + bo
    wt(wo_, kC, kC);
    gemm(y16, 0, 0, kC, wbuf, 0, 0, kC, 0, bo_, h, 0, 0, kC,
         h, 0, 0, kC, nullptr, 0, 0, 0, kBT, kC, kC, 1, 1, 0);
  };

  for (int l = 0; l < kL; ++l) {
    const float* l1w = ln1w + (size_t)l * kC;
    const float* l1b = ln1b + (size_t)l * kC;
    const size_t oCC = (size_t)l * kC * kC;
    const size_t oC  = (size_t)l * kC;

    // self-attention (ln_1)
    ln_f16_kernel<<<kBT, 256, 0, stream>>>(h, l1w, l1b, n16);
    attention(n16, n16, kT, sawq + oCC, sabq + oC, sawk + oCC, sabk + oC,
              sawv + oCC, sabv + oC, sawo + oCC, sabo + oC);

    // cross-attention (ln_1 reused, kv = raw xa, tril mask per reference)
    ln_f16_kernel<<<kBT, 256, 0, stream>>>(h, l1w, l1b, n16);
    attention(n16, xa16, kTA, cawq + oCC, cabq + oC, cawk + oCC, cabk + oC,
              cawv + oCC, cabv + oC, cawo + oCC, cabo + oC);

    // MLP (ln_2 -> fc+gelu -> proj + residual)
    ln_f16_kernel<<<kBT, 256, 0, stream>>>(h, ln2w + oC, ln2b + oC, n16);
    wt(fcw + (size_t)l * kC * kF, kC, kF);
    gemm(n16, 0, 0, kC, wbuf, 0, 0, kC, 0, fcb + (size_t)l * kF,
         nullptr, 0, 0, 0, nullptr, 0, 0, 0,
         m16, 0, 0, kF, kBT, kF, kC, 1, 1, 1);
    wt(prw + (size_t)l * kF * kC, kF, kC);
    gemm(m16, 0, 0, kF, wbuf, 0, 0, kF, 0, prb + oC,
         h, 0, 0, kC, h, 0, 0, kC, nullptr, 0, 0, 0,
         kBT, kC, kF, 1, 1, 0);
  }

  // final LN + tied lm_head: logits = ln_f(h) @ wte^T  (wte [V,C] == Bt layout)
  ln_f16_kernel<<<kBT, 256, 0, stream>>>(h, lnfw, lnfb, n16);
  gemm(n16, 0, 0, kC, wte, 0, 0, kC, /*b_f32=*/1, nullptr,
       nullptr, 0, 0, 0, (float*)d_out, 0, 0, kV, nullptr, 0, 0, 0,
       kBT, kV, kC, 1, 1, 0);
}
